// MLPReceiverBaseline_3934190043769
// MI455X (gfx1250) — compile-verified, matrix operand order checked by reading, not run on hardware
//
#include <hip/hip_runtime.h>

typedef _Float16 v16h __attribute__((ext_vector_type(16)));
typedef float    v8f  __attribute__((ext_vector_type(8)));

union Frag { uint4 u[2]; v16h v; };
union H2U  { _Float16 h[2]; unsigned int u; };

// ---------------- problem constants ----------------
constexpr int B_GRAPHS = 131072;
constexpr int P = 22, F = 14;          // players, features
constexpr int NNODES = B_GRAPHS * P;   // 2,883,584
constexpr int KPAD = 320;              // 308 padded to a multiple of 32 (zeros)
constexpr int H1D = 256, H2D = 128;

// fragment counts (16x16x32 f16 WMMA tiles)
constexpr int W1_NT = 16, W1_KT = 10;  // N=256, K=320
constexpr int W2_NT = 8,  W2_KT = 8;   // N=128, K=256
constexpr int W3_NT = 2,  W3_KT = 4;   // N=32 (22 padded), K=128
constexpr int W1_FRAGS = W1_NT * W1_KT;
constexpr int W2_FRAGS = W2_NT * W2_KT;
constexpr int W3_FRAGS = W3_NT * W3_KT;
constexpr int PACK_TOTAL = (W1_FRAGS + W2_FRAGS + W3_FRAGS) * 512; // 118,784

// ---------------- workspace layout (bytes) ----------------
constexpr size_t STARTS_OFF = 0;
constexpr size_t ENDS_OFF   = (size_t)B_GRAPHS * 4;
constexpr size_t W1P_OFF    = (size_t)B_GRAPHS * 8;       // 1,048,576
constexpr size_t W1P_BYTES  = (size_t)W1_FRAGS * 1024;
constexpr size_t W2P_OFF    = W1P_OFF + W1P_BYTES;
constexpr size_t W2P_BYTES  = (size_t)W2_FRAGS * 1024;    // 65,536
constexpr size_t W3P_OFF    = W2P_OFF + W2P_BYTES;        // contiguous after W2P
constexpr size_t W3P_BYTES  = (size_t)W3_FRAGS * 1024;    // 8,192

// LDS layout inside k_mlp (bytes):
//   [0, 40960)        ldsA [64][320] fp16  (layer 1 only; reused as h2s after)
//   [40960, 73728)    h1s  [64][256] fp16
//   [0, 16384)        h2s  [64][128] fp16  (aliases dead ldsA)
//   [73728, 139264)   w2s  (async-DMA'd copy of w2p)
//   [139264, 147456)  w3s  (async-DMA'd copy of w3p)
constexpr int LDS_A_BYTES  = 64 * KPAD * 2;                 // 40960
constexpr int LDS_H1_BYTES = 64 * H1D * 2;                  // 32768
constexpr int LDS_W2S_OFF  = LDS_A_BYTES + LDS_H1_BYTES;    // 73728
constexpr int LDS_W3S_OFF  = LDS_W2S_OFF + (int)W2P_BYTES;  // 139264
constexpr int LDS_TOTAL    = LDS_W3S_OFF + (int)W3P_BYTES;  // 147456 (144 KB)
constexpr int WCOPY_BYTES  = (int)(W2P_BYTES + W3P_BYTES);  // 73728
constexpr int WCOPY_PER_WV = WCOPY_BYTES / 4;               // 18432
constexpr int WCOPY_ITERS  = WCOPY_PER_WV / 512;            // 36 b128 DMAs per wave

// ---------------- kernels ----------------

__global__ __launch_bounds__(256) void k_init(int* __restrict__ p, int n) {
  int i = blockIdx.x * 256 + threadIdx.x;
  if (i < n) p[i] = 0;
}

// batch is sorted: boundaries give [start, end) node range of each graph.
__global__ __launch_bounds__(256) void k_bounds(const int* __restrict__ batch,
                                                int* __restrict__ starts,
                                                int* __restrict__ ends) {
  int i = blockIdx.x * 256 + threadIdx.x;
  if (i >= NNODES) return;
  int g = batch[i];
  if (i == 0 || batch[i - 1] != g) starts[g] = i;
  if (i == NNODES - 1 || batch[i + 1] != g) ends[g] = i + 1;
}

// Pack W (row-major [Nout, K]) into WMMA B-fragment order:
// element index fe = frag*512 + lane*16 + h ; frag = nt*KT + kt
// K(lane,h) = kt*32 + (h>=8?16:0) + (lane>=16?8:0) + (h&7) ; N = nt*16 + (lane&15)
__global__ __launch_bounds__(256) void k_pack(const float* __restrict__ W1,
                                              const float* __restrict__ W2,
                                              const float* __restrict__ W3,
                                              _Float16* __restrict__ w1p,
                                              _Float16* __restrict__ w2p,
                                              _Float16* __restrict__ w3p) {
  int e = blockIdx.x * 256 + threadIdx.x;
  if (e >= PACK_TOTAL) return;
  const float* src; _Float16* dst; int fe, KT, Kreal, Nreal;
  if (e < W1_FRAGS * 512) {
    fe = e;                               src = W1; dst = w1p; KT = W1_KT; Kreal = 308; Nreal = 256;
  } else if (e < (W1_FRAGS + W2_FRAGS) * 512) {
    fe = e - W1_FRAGS * 512;              src = W2; dst = w2p; KT = W2_KT; Kreal = 256; Nreal = 128;
  } else {
    fe = e - (W1_FRAGS + W2_FRAGS) * 512; src = W3; dst = w3p; KT = W3_KT; Kreal = 128; Nreal = 22;
  }
  int frag = fe >> 9;
  int lane = (fe >> 4) & 31;
  int h    = fe & 15;
  int nt = frag / KT, kt = frag % KT;
  int k = kt * 32 + ((h >= 8) ? 16 : 0) + ((lane >= 16) ? 8 : 0) + (h & 7);
  int n = nt * 16 + (lane & 15);
  float v = (k < Kreal && n < Nreal) ? src[(size_t)n * Kreal + k] : 0.0f;
  dst[fe] = (_Float16)v;
}

// CDNA5 async global->LDS DMA (tracked by ASYNCcnt).
// GVS mode: lds[vdst + 0..15] = mem[saddr + vaddr + 0..15]
__device__ __forceinline__ void async_copy_b128(unsigned lds_off, unsigned gl_off,
                                                const void* sbase) {
  asm volatile("global_load_async_to_lds_b128 %0, %1, %2 offset:0"
               :: "v"(lds_off), "v"(gl_off), "s"(sbase)
               : "memory");
}

// Fused gather + 3-layer MLP. 4 waves/block; each wave owns a 16-row M-tile
// (64 graphs/block). batch is sorted, so the block's source rows of x are one
// contiguous ~79 KB range -> gather straight into an LDS A-tile. W2/W3
// fragment banks are DMA'd into LDS up front (overlapping gather + layer 1).
__global__ __launch_bounds__(128) void k_mlp(const float* __restrict__ x,
                                             const int* __restrict__ starts,
                                             const int* __restrict__ ends,
                                             const _Float16* __restrict__ w1p,
                                             const _Float16* __restrict__ w2p,
                                             const _Float16* __restrict__ w3p,
                                             const float* __restrict__ b1,
                                             const float* __restrict__ b2,
                                             const float* __restrict__ b3,
                                             float* __restrict__ out) {
  __shared__ __align__(16) char ldsbuf[LDS_TOTAL];   // 144 KB
  _Float16* h1s = (_Float16*)(ldsbuf + LDS_A_BYTES);
  _Float16* h2s = (_Float16*)ldsbuf;                 // reuses ldsA region after layer 1

  const int lane = threadIdx.x & 31;
  const int wv   = threadIdx.x >> 5;      // wave 0..3
  const int lhi  = lane >> 4;             // lane-half (0/1)
  const int ln   = lane & 15;
  const int rowBase = blockIdx.x * 64 + wv * 16;

  // ---------- kick off async DMA of W2p+W3p (contiguous in ws) into LDS ----------
  {
    unsigned base = (unsigned)(uintptr_t)(ldsbuf + LDS_W2S_OFF);  // LDS byte offset
    unsigned off0 = wv * WCOPY_PER_WV + lane * 16;
#pragma unroll
    for (int i = 0; i < WCOPY_ITERS; ++i) {
      async_copy_b128(base + off0 + i * 512, off0 + i * 512, (const void*)w2p);
    }
  }

  // ---------------- phase 0: gather/flatten into ldsA[64][320] ----------------
  // 64 rows x 22 player slots = 1408 (row,pos) pairs; each pair moves 14 floats.
#pragma unroll
  for (int p = 0; p < 11; ++p) {
    int pr  = threadIdx.x + p * 128;       // 0..1407
    int r   = pr / P;
    int pos = pr % P;
    int g   = blockIdx.x * 64 + r;
    int s   = starts[g];
    int cnt = ends[g] - s;                 // 0 for empty graphs (k_init)
    unsigned int* dst = (unsigned int*)(ldsbuf + r * (KPAD * 2) + pos * (F * 2));
    if (pos < cnt) {
      const float2* xr = (const float2*)(x + (size_t)(s + pos) * F);
#pragma unroll
      for (int j = 0; j < 7; ++j) {
        float2 v = xr[j];
        H2U cv; cv.h[0] = (_Float16)v.x; cv.h[1] = (_Float16)v.y;
        dst[j] = cv.u;
      }
    } else {
#pragma unroll
      for (int j = 0; j < 7; ++j) dst[j] = 0u;   // truncated / padded slots
    }
  }
  // K padding 308..319 (12 halves per row)
  if (threadIdx.x < 64) {
    unsigned int* dst = (unsigned int*)(ldsbuf + threadIdx.x * (KPAD * 2) + 308 * 2);
#pragma unroll
    for (int j = 0; j < 6; ++j) dst[j] = 0u;
  }
  __syncthreads();

  // ---------------- layer 1: ldsA[64,320] x W1^T[320,256] -> h1s ----------------
  // A-frag (16-bit A 16x32 per ISA): row M = lane&15,
  // halves 0..7 -> K = kt*32 + (lane>=16?8:0) + h, halves 8..15 -> +16.
  const char* aBase = ldsbuf + (wv * 16 + ln) * (KPAD * 2) + lhi * 16;
  Frag a1[W1_KT];
#pragma unroll
  for (int kt = 0; kt < W1_KT; ++kt) {
    a1[kt].u[0] = *(const uint4*)(aBase + kt * 64);
    a1[kt].u[1] = *(const uint4*)(aBase + kt * 64 + 32);
  }
  for (int nt = 0; nt < W1_NT; ++nt) {
    v8f acc = {};
    const char* bp = (const char*)w1p + (size_t)(nt * W1_KT) * 1024 + lane * 32;
#pragma unroll
    for (int kt = 0; kt < W1_KT; ++kt) {
      Frag b;
      b.u[0] = *(const uint4*)(bp + kt * 1024);
      b.u[1] = *(const uint4*)(bp + kt * 1024 + 16);
      acc = __builtin_amdgcn_wmma_f32_16x16x32_f16(false, a1[kt].v, false, b.v,
                                                   (short)0, acc, false, false);
    }
    float bias = b1[nt * 16 + ln];
#pragma unroll
    for (int r = 0; r < 8; ++r) {
      float h = acc[r] + bias;
      h = h > 0.0f ? h : 0.0f;                       // relu
      int M = r + lhi * 8;                           // C layout: M = r + 8*(lane>=16)
      h1s[(wv * 16 + M) * H1D + nt * 16 + ln] = (_Float16)h;
    }
  }
  // W2/W3 DMA must be complete and visible to all waves before layer 2.
  asm volatile("s_wait_asynccnt 0x0" ::: "memory");
  __syncthreads();   // also retires layer-1; ldsA region now reusable as h2s

  // ---------------- layer 2: h1[64,256] x W2^T[256,128] -> h2s ----------------
  const char* a2Base = (const char*)h1s + (wv * 16 + ln) * (H1D * 2) + lhi * 16;
  Frag a2[W2_KT];
#pragma unroll
  for (int kt = 0; kt < W2_KT; ++kt) {
    a2[kt].u[0] = *(const uint4*)(a2Base + kt * 64);
    a2[kt].u[1] = *(const uint4*)(a2Base + kt * 64 + 32);
  }
  for (int nt = 0; nt < W2_NT; ++nt) {
    v8f acc = {};
    const char* bp = ldsbuf + LDS_W2S_OFF + (nt * W2_KT) * 1024 + lane * 32;
#pragma unroll
    for (int kt = 0; kt < W2_KT; ++kt) {
      Frag b;
      b.u[0] = *(const uint4*)(bp + kt * 1024);
      b.u[1] = *(const uint4*)(bp + kt * 1024 + 16);
      acc = __builtin_amdgcn_wmma_f32_16x16x32_f16(false, a2[kt].v, false, b.v,
                                                   (short)0, acc, false, false);
    }
    float bias = b2[nt * 16 + ln];
#pragma unroll
    for (int r = 0; r < 8; ++r) {
      float h = acc[r] + bias;
      h = h > 0.0f ? h : 0.0f;
      int M = r + lhi * 8;
      h2s[(wv * 16 + M) * H2D + nt * 16 + ln] = (_Float16)h;
    }
  }
  __syncthreads();

  // ---------------- layer 3: h2[64,128] x W3^T[128,22] -> logits ----------------
  const char* a3Base = (const char*)h2s + (wv * 16 + ln) * (H2D * 2) + lhi * 16;
  Frag a3[W3_KT];
#pragma unroll
  for (int kt = 0; kt < W3_KT; ++kt) {
    a3[kt].u[0] = *(const uint4*)(a3Base + kt * 64);
    a3[kt].u[1] = *(const uint4*)(a3Base + kt * 64 + 32);
  }
#pragma unroll
  for (int nt = 0; nt < W3_NT; ++nt) {
    v8f acc = {};
    const char* bp = ldsbuf + LDS_W3S_OFF + (nt * W3_KT) * 1024 + lane * 32;
#pragma unroll
    for (int kt = 0; kt < W3_KT; ++kt) {
      Frag b;
      b.u[0] = *(const uint4*)(bp + kt * 1024);
      b.u[1] = *(const uint4*)(bp + kt * 1024 + 16);
      acc = __builtin_amdgcn_wmma_f32_16x16x32_f16(false, a3[kt].v, false, b.v,
                                                   (short)0, acc, false, false);
    }
    int n = nt * 16 + ln;
    if (n < P) {
      float bias = b3[n];
#pragma unroll
      for (int r = 0; r < 8; ++r) {
        int M = r + lhi * 8;
        out[(size_t)(rowBase + M) * P + n] = acc[r] + bias;
      }
    }
  }
}

// ---------------- launch ----------------
extern "C" void kernel_launch(void* const* d_in, const int* in_sizes, int n_in,
                              void* d_out, int out_size, void* d_ws, size_t ws_size,
                              hipStream_t stream) {
  const float* x    = (const float*)d_in[0];
  const int*   batch= (const int*)  d_in[1];
  const float* W1   = (const float*)d_in[2];
  const float* b1   = (const float*)d_in[3];
  const float* W2   = (const float*)d_in[4];
  const float* b2   = (const float*)d_in[5];
  const float* W3   = (const float*)d_in[6];
  const float* b3   = (const float*)d_in[7];
  float* out = (float*)d_out;

  char* ws = (char*)d_ws;
  int*      starts = (int*)     (ws + STARTS_OFF);
  int*      ends   = (int*)     (ws + ENDS_OFF);
  _Float16* w1p    = (_Float16*)(ws + W1P_OFF);
  _Float16* w2p    = (_Float16*)(ws + W2P_OFF);
  _Float16* w3p    = (_Float16*)(ws + W3P_OFF);

  k_init  <<<(2 * B_GRAPHS) / 256, 256, 0, stream>>>(starts, 2 * B_GRAPHS);
  k_bounds<<<NNODES / 256, 256, 0, stream>>>(batch, starts, ends);
  k_pack  <<<(PACK_TOTAL + 255) / 256, 256, 0, stream>>>(W1, W2, W3, w1p, w2p, w3p);
  k_mlp   <<<B_GRAPHS / 64, 128, 0, stream>>>(x, starts, ends, w1p, w2p, w3p,
                                              b1, b2, b3, out);
}
